// GCNModelVAE_62843961475769
// MI455X (gfx1250) — matrix operand order repaired and checked by reading, not compiled, not run on hardware
//
#include <hip/hip_runtime.h>
#include <hip/hip_bf16.h>

// ---------------------------------------------------------------------------
// GCN-VAE forward for MI455X (gfx1250, wave32, WMMA + TDM).
// All GEMMs run through v_wmma_f32_16x16x32_f16 (f16 operands, f32 accum).
// Classifier stages mu tiles into LDS with tensor_load_to_lds (TENSORcnt).
// ---------------------------------------------------------------------------

typedef __attribute__((ext_vector_type(16))) _Float16 v16h;
typedef __attribute__((ext_vector_type(8)))  _Float16 v8h;
typedef __attribute__((ext_vector_type(8)))  float    v8f;
typedef __attribute__((ext_vector_type(4)))  float    v4f;
typedef __attribute__((ext_vector_type(4)))  unsigned int v4u;
typedef __attribute__((ext_vector_type(8)))  int      v8i;
typedef __attribute__((ext_vector_type(4)))  int      v4i;

static constexpr int NN = 1024;   // nodes
static constexpr int FF = 256;    // input features
static constexpr int HH = 64;     // hidden
static constexpr int EE = 32768;  // edges

#if defined(__has_builtin)
#if __has_builtin(__builtin_amdgcn_tensor_load_to_lds)
#define HAS_TDM 1
#endif
#endif
#ifndef HAS_TDM
#define HAS_TDM 0
#endif

__device__ inline v8f wmma_f16(v16h a, v16h b, v8f c) {
  return __builtin_amdgcn_wmma_f32_16x16x32_f16(
      /*neg_a=*/false, a, /*neg_b=*/false, b,
      /*c_mod=*/(short)0, c, /*reuse_a=*/false, /*reuse_b=*/false);
}

#if HAS_TDM
// TDM: DMA a 16-row x 64-half tile of an f16 [1024, 64] row-major tensor
// into LDS. D# bitfields per CDNA5 ISA 8.3/8.4 (2D tensor, groups 2/3 zero).
__device__ inline void tdm_load_tile16x64(const _Float16* gsrc, void* lds_dst) {
  unsigned long long ga = (unsigned long long)(uintptr_t)gsrc;
  unsigned int lds_addr = (unsigned int)(uintptr_t)lds_dst;  // low 32 = LDS byte offset
  v4u g0;
  g0[0] = 1u;                                   // count=1, user mode
  g0[1] = lds_addr;                             // lds_addr[31:0]
  g0[2] = (unsigned int)ga;                     // global_addr[31:0]
  g0[3] = (unsigned int)((ga >> 32) & 0x1FFFFFFu) | (2u << 30);  // addr[56:32]|type=2
  v8i g1;
  g1[0] = (1 << 16);          // workgroup_mask=0, data_size=1 (2 bytes)
  g1[1] = (64 << 16);         // tensor_dim0[15:0] = 64   (bits 79:48)
  g1[2] = (int)(1024u << 16); // tensor_dim0 hi=0 | tensor_dim1[15:0] = 1024
  g1[3] = (64 << 16);         // tensor_dim1 hi=0 | tile_dim0 = 64
  g1[4] = 16;                 // tile_dim1 = 16, tile_dim2 = 0
  g1[5] = 64;                 // tensor_dim0_stride = 64  (bits 207:160)
  g1[6] = 0;                  // tensor_dim0_stride hi | tensor_dim1_stride lo
  g1[7] = 0;                  // tensor_dim1_stride hi
  v4i z = {0, 0, 0, 0};
#if __clang_major__ >= 23
  v8i z8 = {0, 0, 0, 0, 0, 0, 0, 0};
  __builtin_amdgcn_tensor_load_to_lds(g0, g1, z, z, z8, 0);
#else
  __builtin_amdgcn_tensor_load_to_lds(g0, g1, z, z, 0);
#endif
}
#endif

// A operand (16x32 MxK per wmma) loaded from an f32 row with conversion.
__device__ inline v16h load_a_f32(const float* __restrict__ row, int kbase) {
  const v4f* p0 = (const v4f*)(row + kbase);
  const v4f* p1 = (const v4f*)(row + kbase + 8);
  v4f c0 = p0[0], c1 = p0[1], c2 = p1[0], c3 = p1[1];
  v16h r;
#pragma unroll
  for (int i = 0; i < 4; ++i) {
    r[i]      = (_Float16)c0[i];
    r[4 + i]  = (_Float16)c1[i];
    r[8 + i]  = (_Float16)c2[i];
    r[12 + i] = (_Float16)c3[i];
  }
  return r;
}

// A operand loaded from an f16 row (two contiguous 8-half chunks).
__device__ inline v16h load_a_f16(const _Float16* row, int kbase) {
  union { v16h v; v8h h[2]; } u;
  u.h[0] = *(const v8h*)(row + kbase);
  u.h[1] = *(const v8h*)(row + kbase + 8);
  return u.v;
}

// B operand (32x16 KxN) from pre-packed layout: 16 contiguous halfs per lane.
__device__ inline v16h load_b_pack(const _Float16* __restrict__ pack,
                                   int ks, int g, int Gn, int lane) {
  const _Float16* p = pack + (((size_t)(ks * Gn + g) * 32 + lane) << 4);
  union { v16h v; v8h h[2]; } u;
  u.h[0] = ((const v8h*)p)[0];
  u.h[1] = ((const v8h*)p)[1];
  return u.v;
}

// B operand built from 16 contiguous f16 values.
__device__ inline v16h load_b_f16(const _Float16* p) {
  union { v16h v; v8h h[2]; } u;
  u.h[0] = ((const v8h*)p)[0];
  u.h[1] = ((const v8h*)p)[1];
  return u.v;
}

// ------------------------------- prep kernels ------------------------------

__global__ void k_fill(float* __restrict__ p, int n, float v) {
  int i = blockIdx.x * blockDim.x + threadIdx.x;
  if (i < n) p[i] = v;
}

__global__ void k_deg_count(float* __restrict__ deg, const int* __restrict__ dst, int e) {
  int i = blockIdx.x * blockDim.x + threadIdx.x;
  if (i < e) atomicAdd(&deg[dst[i]], 1.0f);
}

__global__ void k_dis(float* __restrict__ dis, const float* __restrict__ deg, int n) {
  int i = blockIdx.x * blockDim.x + threadIdx.x;
  if (i < n) dis[i] = rsqrtf(deg[i]);
}

__global__ void k_sum_wc1(float* __restrict__ s, const float* __restrict__ Wc1) {
  int j = threadIdx.x;  // 64 threads
  s[j] = Wc1[j] + Wc1[64 + j] + Wc1[128 + j] + Wc1[192 + j];
}

// Pack an f32 [K, Nlog] row-major matrix into a contiguous-k WMMA B-operand layout
// (f16), zero-padding columns up to Npad. Slot t = ((ks*Gn+g)*32+lane)*16 halfs:
//   N = 16*g + (lane&15),  K = 32*ks + (lane&16 ? 16 : 0) + e,  e = 0..15
__global__ void k_pack_b(const float* __restrict__ src, _Float16* __restrict__ dst,
                         int Klog, int Nlog, int Npad) {
  int t = blockIdx.x * blockDim.x + threadIdx.x;
  int Gn = Npad >> 4;
  int total = (Klog >> 5) * Gn * 32;
  if (t >= total) return;
  int lane = t & 31;
  int rest = t >> 5;
  int g  = rest % Gn;
  int ks = rest / Gn;
  int N  = 16 * g + (lane & 15);
  int kb = 32 * ks + ((lane & 16) ? 16 : 0);
  _Float16* d = dst + (size_t)t * 16;
#pragma unroll
  for (int e = 0; e < 16; ++e) {
    float v = (N < Nlog) ? src[(size_t)(kb + e) * Nlog + N] : 0.0f;
    d[e] = (_Float16)v;
  }
}

// ------------------------------- WMMA GEMM --------------------------------
// C[M,Nout] = A[M,Kdim] (f32, row-major, converted on the fly) @ Bpack.
// One 16x16 output tile per wave32. grid = (M/16)*Gn blocks of 32 threads.
__global__ __launch_bounds__(32) void k_gemm(const float* __restrict__ A,
                                             const _Float16* __restrict__ Bp,
                                             float* __restrict__ C,
                                             int Kdim, int Gn, int Nout) {
  int lane = threadIdx.x;
  int nt = blockIdx.x % Gn;
  int mt = blockIdx.x / Gn;
  int l15 = lane & 15;
  int laneoff = (lane & 16) ? 16 : 0;
  const float* arow = A + (size_t)(mt * 16 + l15) * Kdim;
  v8f acc = {};
  int KS = Kdim >> 5;
  for (int ks = 0; ks < KS; ++ks) {
    if (ks + 1 < KS) __builtin_prefetch(arow + 32 * (ks + 1) + laneoff, 0, 3);
    v16h a = load_a_f32(arow, 32 * ks + laneoff);
    v16h b = load_b_pack(Bp, ks, nt, Gn, lane);
    acc = wmma_f16(a, b, acc);
  }
  int col   = nt * 16 + l15;
  int rbase = mt * 16 + ((lane & 16) ? 8 : 0);
#pragma unroll
  for (int r = 0; r < 8; ++r) C[(size_t)(rbase + r) * Nout + col] = acc[r];
}

// ------------------------- GCN aggregation kernels -------------------------

// out[n,j] = h[n,j] * dis[n]^2 + bias[j]     (self-loop term + bias)
__global__ void k_agg_init(float* __restrict__ out, const float* __restrict__ h,
                           const float* __restrict__ dis, const float* __restrict__ bias) {
  int i = blockIdx.x * blockDim.x + threadIdx.x;  // N*H threads
  if (i >= NN * HH) return;
  int n = i >> 6, j = i & 63;
  out[i] = h[i] * dis[n] * dis[n] + bias[j];
}

// out[dst,j] += h[src,j] * dis[src]*dis[dst]   (per-edge message, atomics)
__global__ void k_agg_scatter(float* __restrict__ out, const float* __restrict__ h,
                              const float* __restrict__ dis,
                              const int* __restrict__ src, const int* __restrict__ dst) {
  int i = blockIdx.x * blockDim.x + threadIdx.x;  // E*H threads
  if (i >= EE * HH) return;
  int e = i >> 6, j = i & 63;
  int s = src[e], d = dst[e];
  atomicAdd(&out[(size_t)d * HH + j], h[(size_t)s * HH + j] * dis[s] * dis[d]);
}

__global__ void k_cvt_f16(_Float16* __restrict__ dstp, const float* __restrict__ srcp, int n) {
  int i = blockIdx.x * blockDim.x + threadIdx.x;
  if (i < n) dstp[i] = (_Float16)srcp[i];
}

// ------------------------- fused classifier kernel -------------------------
// One wave per 16x16 (n,m) node tile:
//   TDM: mu row-tiles (n_base.., m_base..) -> LDS     (tensor_load_to_lds)
//   adj = sigmoid(mu_n . mu_m)                        (2 WMMAs)
//   for each of 16 n-rows: h1=relu(a*s+b1) -> @Wc2 (8 WMMAs) -> relu
//     -> LDS transpose -> @Wc3pad (2 WMMAs) -> sigmoid -> store 4 classes
__global__ __launch_bounds__(32) void k_classifier(
    const _Float16* __restrict__ muh, const _Float16* __restrict__ B2p,
    const _Float16* __restrict__ B3p, const float* __restrict__ svec,
    const float* __restrict__ bc1, const float* __restrict__ bc2,
    const float* __restrict__ bc3, float* __restrict__ out) {
  int lane = threadIdx.x;
  int nt = blockIdx.x >> 6;      // 64 m-tiles per n-tile
  int mt = blockIdx.x & 63;
  int n_base = nt * 16, m_base = mt * 16;
  int l15 = lane & 15;
  bool hi = (lane & 16) != 0;
  int laneoffA = hi ? 8 : 0;
  int laneoffB = hi ? 16 : 0;

  __shared__ __align__(16) float aT[256];
  __shared__ __align__(16) _Float16 stage[16 * 64];
  __shared__ float sS[64];
  __shared__ float sB1[64];
#if HAS_TDM
  __shared__ __align__(16) _Float16 muA[16 * 64];
  __shared__ __align__(16) _Float16 muB[16 * 64];
  // kick off the DMA of both mu tiles before doing anything else
  tdm_load_tile16x64(muh + (size_t)n_base * HH, (void*)muA);
  tdm_load_tile16x64(muh + (size_t)m_base * HH, (void*)muB);
#endif

  sS[lane] = svec[lane];        sS[lane + 32] = svec[lane + 32];
  sB1[lane] = bc1[lane];        sB1[lane + 32] = bc1[lane + 32];
  __syncthreads();

  // per-lane biases for the C-layout of the big GEMM
  float b2g[4];
#pragma unroll
  for (int g = 0; g < 4; ++g) b2g[g] = bc2[l15 + 16 * g];
  float b3 = (l15 < 4) ? bc3[l15] : 0.0f;

  // per-lane slice of s / bc1 for the A (h1) operand, in f16
  _Float16 sh[32], bh[32];
#pragma unroll
  for (int ks = 0; ks < 2; ++ks)
#pragma unroll
    for (int e = 0; e < 16; ++e) {
      int K = 32 * ks + laneoffB + e;
      sh[ks * 16 + e] = (_Float16)sS[K];
      bh[ks * 16 + e] = (_Float16)sB1[K];
    }

  // register-resident B operands: Wc2 (64x64) and Wc3 (64x16, zero-padded)
  v16h B2[2][4], B3[2];
#pragma unroll
  for (int ks = 0; ks < 2; ++ks) {
#pragma unroll
    for (int g = 0; g < 4; ++g) B2[ks][g] = load_b_pack(B2p, ks, g, 4, lane);
    B3[ks] = load_b_pack(B3p, ks, 0, 1, lane);
  }

  // ---- adj tile: sigmoid(mu[n_base+i] . mu[m_base+j]) via WMMA ----
#if HAS_TDM
  __builtin_amdgcn_s_wait_tensorcnt(0);   // both tiles landed in LDS
  const _Float16* arow = muA + (size_t)l15 * HH;
  const _Float16* brow = muB + (size_t)l15 * HH;
#else
  const _Float16* arow = muh + (size_t)(n_base + l15) * HH;
  const _Float16* brow = muh + (size_t)(m_base + l15) * HH;
#endif
  v8f cadj = {};
#pragma unroll
  for (int ks = 0; ks < 2; ++ks) {
    v16h a = load_a_f16(arow, 32 * ks + laneoffB);
    v16h b = load_b_f16(brow + 32 * ks + laneoffB);
    cadj = wmma_f16(a, b, cadj);
  }
#pragma unroll
  for (int r = 0; r < 8; ++r) {
    int nl = r + (hi ? 8 : 0);
    aT[nl * 16 + l15] = 1.0f / (1.0f + __expf(-cadj[r]));
  }
  __syncthreads();

  // ---- 16 sub-rows: each handles (n = n_base+sub, all 16 m) ----
  for (int sub = 0; sub < 16; ++sub) {
    float a = aT[sub * 16 + l15];   // lane handles M-row = l15 (pair m-index)
    _Float16 ah = (_Float16)a;

    // h1 = relu(a*s + b1) packed directly into the A layout
    v16h A1[2];
#pragma unroll
    for (int ks = 0; ks < 2; ++ks)
#pragma unroll
      for (int e = 0; e < 16; ++e) {
        _Float16 v = ah * sh[ks * 16 + e] + bh[ks * 16 + e];
        A1[ks][e] = (v > (_Float16)0) ? v : (_Float16)0;
      }

    v8f h2[4] = {{}, {}, {}, {}};
#pragma unroll
    for (int g = 0; g < 4; ++g) {
      h2[g] = wmma_f16(A1[0], B2[0][g], h2[g]);
      h2[g] = wmma_f16(A1[1], B2[1][g], h2[g]);
    }

    // relu(+bc2), transpose C-layout -> A-layout through LDS
    __syncthreads();
#pragma unroll
    for (int g = 0; g < 4; ++g)
#pragma unroll
      for (int r = 0; r < 8; ++r) {
        float v = h2[g][r] + b2g[g];
        v = (v > 0.0f) ? v : 0.0f;
        stage[(r + (hi ? 8 : 0)) * 64 + 16 * g + l15] = (_Float16)v;
      }
    __syncthreads();

    v8f c3 = {};
#pragma unroll
    for (int ks = 0; ks < 2; ++ks) {
      union { v16h v; v8h h[2]; } ua;
      ua.h[0] = *(const v8h*)&stage[l15 * 64 + 32 * ks + laneoffB];
      ua.h[1] = *(const v8h*)&stage[l15 * 64 + 32 * ks + laneoffB + 8];
      c3 = wmma_f16(ua.v, B3[ks], c3);
    }

    if (l15 < 4) {  // only class columns 0..3 are real
      int n = n_base + sub;
#pragma unroll
      for (int r = 0; r < 8; ++r) {
        int m = m_base + r + (hi ? 8 : 0);
        out[((size_t)n * NN + m) * 4 + l15] = 1.0f / (1.0f + __expf(-(c3[r] + b3)));
      }
    }
  }
}

// ------------------------------- host driver -------------------------------

extern "C" void kernel_launch(void* const* d_in, const int* in_sizes, int n_in,
                              void* d_out, int out_size, void* d_ws, size_t ws_size,
                              hipStream_t stream) {
  const float* x   = (const float*)d_in[0];
  const int*   src = (const int*)d_in[1];
  const int*   dst = (const int*)d_in[2];
  const float* W1  = (const float*)d_in[3];
  const float* b1  = (const float*)d_in[4];
  const float* W2  = (const float*)d_in[5];
  const float* b2  = (const float*)d_in[6];
  const float* W3  = (const float*)d_in[7];
  const float* b3  = (const float*)d_in[8];
  const float* Wc1 = (const float*)d_in[9];
  const float* bc1 = (const float*)d_in[10];
  const float* Wc2 = (const float*)d_in[11];
  const float* bc2 = (const float*)d_in[12];
  const float* Wc3 = (const float*)d_in[13];
  const float* bc3 = (const float*)d_in[14];
  float* out = (float*)d_out;

  // output layout: a_hat [N*N*4] | mu [N*H] | logvar [N*H]
  const size_t OUT_MU = (size_t)NN * NN * 4;
  const size_t OUT_LV = OUT_MU + (size_t)NN * HH;

  // workspace layout (byte offsets)
  char* w = (char*)d_ws;
  float*    deg    = (float*)(w + 0);              // 4 KB
  float*    dis    = (float*)(w + (4 << 10));      // 4 KB
  _Float16* pW1    = (_Float16*)(w + (8 << 10));   // 32 KB
  _Float16* pW2    = (_Float16*)(w + (40 << 10));  // 8 KB
  _Float16* pW3    = (_Float16*)(w + (48 << 10));  // 8 KB
  _Float16* pWc2   = (_Float16*)(w + (56 << 10));  // 8 KB
  _Float16* pWc3   = (_Float16*)(w + (64 << 10));  // 2 KB
  float*    svec   = (float*)(w + (66 << 10));     // 256 B
  float*    htmp   = (float*)(w + (68 << 10));     // 256 KB
  float*    z1     = (float*)(w + (324 << 10));    // 256 KB
  float*    hmu    = (float*)(w + (580 << 10));    // 256 KB
  float*    hlv    = (float*)(w + (836 << 10));    // 256 KB
  _Float16* muh    = (_Float16*)(w + (1092 << 10)); // 128 KB

  // degree / normalization
  k_fill<<<(NN + 255) / 256, 256, 0, stream>>>(deg, NN, 1.0f);
  k_deg_count<<<(EE + 255) / 256, 256, 0, stream>>>(deg, dst, EE);
  k_dis<<<(NN + 255) / 256, 256, 0, stream>>>(dis, deg, NN);
  k_sum_wc1<<<1, 64, 0, stream>>>(svec, Wc1);

  // pack all weight matrices into WMMA B layout (f16)
  k_pack_b<<<4, 256, 0, stream>>>(W1,  pW1,  FF, HH, HH);
  k_pack_b<<<1, 256, 0, stream>>>(W2,  pW2,  HH, HH, HH);
  k_pack_b<<<1, 256, 0, stream>>>(W3,  pW3,  HH, HH, HH);
  k_pack_b<<<1, 256, 0, stream>>>(Wc2, pWc2, HH, HH, HH);
  k_pack_b<<<1, 64,  0, stream>>>(Wc3, pWc3, HH, 4, 16);

  // GCN layer 1: z1 = norm-agg(x @ W1) + b1
  k_gemm<<<(NN / 16) * (HH / 16), 32, 0, stream>>>(x, pW1, htmp, FF, HH / 16, HH);
  k_agg_init<<<(NN * HH + 255) / 256, 256, 0, stream>>>(z1, htmp, dis, b1);
  k_agg_scatter<<<(EE * HH + 255) / 256, 256, 0, stream>>>(z1, htmp, dis, src, dst);

  // GCN layers 2/3: mu, logvar (written straight to d_out)
  k_gemm<<<(NN / 16) * (HH / 16), 32, 0, stream>>>(z1, pW2, hmu, HH, HH / 16, HH);
  k_gemm<<<(NN / 16) * (HH / 16), 32, 0, stream>>>(z1, pW3, hlv, HH, HH / 16, HH);
  k_agg_init<<<(NN * HH + 255) / 256, 256, 0, stream>>>(out + OUT_MU, hmu, dis, b2);
  k_agg_scatter<<<(EE * HH + 255) / 256, 256, 0, stream>>>(out + OUT_MU, hmu, dis, src, dst);
  k_agg_init<<<(NN * HH + 255) / 256, 256, 0, stream>>>(out + OUT_LV, hlv, dis, b3);
  k_agg_scatter<<<(EE * HH + 255) / 256, 256, 0, stream>>>(out + OUT_LV, hlv, dis, src, dst);

  // mu -> f16 copy for the WMMA classifier
  k_cvt_f16<<<(NN * HH + 255) / 256, 256, 0, stream>>>(muh, out + OUT_MU, NN * HH);

  // fused decoder + classifier: 4096 node tiles, one wave each
  k_classifier<<<(NN / 16) * (NN / 16), 32, 0, stream>>>(muh, pWc2, pWc3, svec,
                                                         bc1, bc2, bc3, out);
}
